// unit_gcn_3d_4415226380409
// MI455X (gfx1250) — matrix operand order repaired and checked by reading, not compiled, not run on hardware
//
#include <hip/hip_runtime.h>

// ---------------------------------------------------------------------------
// unit_gcn (attention GCN) for MI455X / gfx1250, bf16 WMMA path, v3.
// vs v2: q/k LDS layout switched to [v][t*16+o] (same phase-2 operand
// addresses, consistent kappa remap) so all result write-backs become single
// 16-byte vector stores (ds_store_b128 / global_store_b128).
// ---------------------------------------------------------------------------

#define NS   3
#define CCH  64
#define T2   32
#define V2   75
#define VP   80   // padded joint dim (5 x 16)
#define KP   96   // padded attention K rows (3 x 32)

typedef __bf16 v16bf __attribute__((ext_vector_type(16)));
typedef __bf16 v8bf  __attribute__((ext_vector_type(8)));
typedef float  v8f   __attribute__((ext_vector_type(8)));

__device__ __forceinline__ v8f wmma_bf16(v16bf a, v16bf b, v8f c) {
  // D = A(16x32 bf16) x B(32x16 bf16) + C(16x16 f32)
  return __builtin_amdgcn_wmma_f32_16x16x32_bf16(false, a, false, b, (short)0, c,
                                                 false, false);
}
__device__ __forceinline__ v16bf bfcat(v8bf lo, v8bf hi) {
  return __builtin_shufflevector(lo, hi, 0, 1, 2, 3, 4, 5, 6, 7, 8, 9, 10, 11,
                                 12, 13, 14, 15);
}
__device__ __forceinline__ v8bf ldv8(const __bf16* p) {
  return *(const v8bf*)p;  // 16B contiguous load
}
__device__ __forceinline__ void stv8(__bf16* p, v8bf v) {
  *(v8bf*)p = v;           // 16B contiguous store
}
__device__ __forceinline__ float sigm(float v) { return 1.0f / (1.0f + expf(-v)); }

// ---------------------------------------------------------------------------
// Kernel A: per (n, subset): att = PA + tanh((q^T k)/512)*alpha, stored
// TRANSPOSED + zero padded as [u:80][k:96] bf16 in workspace.
// q/k in LDS as [v][kappa], kappa = t_local*16 + o (contiguous write-back).
// 5 waves / block; wave w owns attention rows [16w,16w+16).
// ---------------------------------------------------------------------------
__global__ __launch_bounds__(160) void attn_kernel(
    const float* __restrict__ x, const float* __restrict__ PA,
    const float* __restrict__ alpha, const float* __restrict__ Wq,
    const float* __restrict__ bq, const float* __restrict__ Wk,
    const float* __restrict__ bk, __bf16* __restrict__ attw) {
  __shared__ __attribute__((aligned(16))) __bf16 qs2[VP][128];  // 20 KB
  __shared__ __attribute__((aligned(16))) __bf16 ks2[VP][128];  // 20 KB
  const int n    = blockIdx.x / NS;
  const int i    = blockIdx.x - n * NS;
  const int tid  = threadIdx.x;
  const int lane = tid & 31;
  const int wave = tid >> 5;   // 0..4
  const int lhi  = lane >> 4;  // 0/1
  const int l16  = lane & 15;
  const float* xb = x + (size_t)n * (CCH * 2400);

  // Projection A-operands (Wq/Wk rows) and biases, kept in VGPRs.
  v16bf aq[2], ak[2];
#pragma unroll
  for (int ch = 0; ch < 2; ++ch)
#pragma unroll
    for (int e = 0; e < 16; ++e) {
      int k = 32 * ch + ((e >> 3) << 4) + (lhi << 3) + (e & 7);
      aq[ch][e] = (__bf16)Wq[(i * 16 + l16) * 64 + k];
      ak[ch][e] = (__bf16)Wk[(i * 16 + l16) * 64 + k];
    }
  float bqv[8], bkv[8];
#pragma unroll
  for (int r = 0; r < 8; ++r) {
    bqv[r] = bq[i * 16 + r + 8 * lhi];
    bkv[r] = bk[i * 16 + r + 8 * lhi];
  }

  v8f cAtt[5];
#pragma unroll
  for (int nt = 0; nt < 5; ++nt)
#pragma unroll
    for (int r = 0; r < 8; ++r) cAtt[nt][r] = 0.0f;

  for (int qt = 0; qt < 4; ++qt) {  // quarters of t (8 each)
    // ---- phase 1: q,k projections for t in [8qt, 8qt+8) ----
    for (int nt = wave; nt < 40; nt += 5) {  // 8 t-slices x 5 v-groups
      const int tl  = nt / 5;
      const int g   = nt - tl * 5;
      const int t   = qt * 8 + tl;
      const int v   = g * 16 + l16;
      const bool ok = (v < V2);
      const float* xp = xb + t * V2 + (ok ? v : 0);  // clamped, always valid
      v8f cq, ck2;
#pragma unroll
      for (int r = 0; r < 8; ++r) { cq[r] = 0.0f; ck2[r] = 0.0f; }
#pragma unroll
      for (int ch = 0; ch < 2; ++ch) {
        v16bf bx;  // B[k=c][n=(t,v)]
#pragma unroll
        for (int e = 0; e < 16; ++e) {
          int c = 32 * ch + ((e >> 3) << 4) + (lhi << 3) + (e & 7);
          float xv = xp[c * 2400];            // unconditional load
          bx[e] = (__bf16)(ok ? xv : 0.0f);   // v_cndmask, no branch
        }
        cq  = wmma_bf16(aq[ch], bx, cq);
        ck2 = wmma_bf16(ak[ch], bx, ck2);
      }
      // kappa = tl*16 + o; consecutive r -> contiguous -> one b128 store each
      v8bf pq, pk;
#pragma unroll
      for (int r = 0; r < 8; ++r) {
        pq[r] = (__bf16)(cq[r]  + bqv[r]);
        pk[r] = (__bf16)(ck2[r] + bkv[r]);
      }
      stv8(&qs2[v][tl * 16 + 8 * lhi], pq);
      stv8(&ks2[v][tl * 16 + 8 * lhi], pk);
    }
    __syncthreads();
    // ---- phase 2: att += qf^T kf (K=32 chunks), vector LDS loads ----------
#pragma unroll
    for (int ch = 0; ch < 4; ++ch) {
      const int ab = ch * 32 + 16 * lhi;  // A: one run of 16
      v16bf A = bfcat(ldv8(&qs2[16 * wave + l16][ab]),
                      ldv8(&qs2[16 * wave + l16][ab + 8]));
      const int bb = ch * 32 + 16 * lhi;  // B: one run of 16
#pragma unroll
      for (int nt = 0; nt < 5; ++nt) {
        v16bf Bt = bfcat(ldv8(&ks2[16 * nt + l16][bb]),
                         ldv8(&ks2[16 * nt + l16][bb + 8]));
        cAtt[nt] = wmma_bf16(A, Bt, cAtt[nt]);
      }
    }
    __syncthreads();
  }

  // Epilogue: att = PA + tanh(att/512)*alpha, store TRANSPOSED [u][k] bf16,
  // one 16B global store per N tile.
  const float al = alpha[0];
  __bf16* abt = attw + (size_t)(n * NS + i) * (VP * KP);
#pragma unroll
  for (int nt = 0; nt < 5; ++nt) {
    const int u   = 16 * nt + l16;
    const int ucl = (u < V2) ? u : 0;
    v8bf pv;
#pragma unroll
    for (int r = 0; r < 8; ++r) {
      const int v   = 16 * wave + r + 8 * lhi;
      const int vcl = (v < V2) ? v : 0;
      float pa  = PA[(i * V2 + vcl) * V2 + ucl];  // clamped, unconditional
      float val = pa + tanhf(cAtt[nt][r] * (1.0f / 512.0f)) * al;
      pv[r] = (__bf16)((v < V2 && u < V2) ? val : 0.0f);
    }
    stv8(&abt[u * KP + 16 * wave + 8 * lhi], pv);
  }
  for (int idx = tid; idx < VP * (KP - VP); idx += 160) {  // zero cols 80..95
    int u = idx >> 4, j = idx & 15;
    abt[u * KP + VP + j] = (__bf16)0.0f;
  }
}

// ---------------------------------------------------------------------------
// Kernel B: per (n, t): V_i = X_t @ att_i (stacked, transposed in LDS), then
// y = Wffn_big(64x192) @ V + bias, BN(eval), +residual, ReLU  ->  d_out.
// 4 waves / block; all WMMA operands are contiguous 16B vector loads.
// ---------------------------------------------------------------------------
__global__ __launch_bounds__(128) void gcn_kernel(
    const float* __restrict__ x, const __bf16* __restrict__ attw,
    const float* __restrict__ Wffn, const float* __restrict__ bffn,
    const float* __restrict__ bn_g, const float* __restrict__ bn_b,
    const float* __restrict__ bn_m, const float* __restrict__ bn_v,
    float* __restrict__ y) {
  __shared__ __attribute__((aligned(16))) __bf16 Xs[CCH][KP];        // 12 KB
  __shared__ __attribute__((aligned(16))) __bf16 Vst[VP][NS * CCH];  // 30 KB
  const int n    = blockIdx.x >> 5;
  const int t    = blockIdx.x & 31;
  const int tid  = threadIdx.x;
  const int lane = tid & 31;
  const int wave = tid >> 5;   // 0..3
  const int lhi  = lane >> 4;
  const int l16  = lane & 15;
  const float* xb = x + (size_t)n * (CCH * 2400) + t * V2;

  for (int idx = tid; idx < CCH * KP; idx += 128) {
    int c = idx / KP, v = idx - c * KP;
    float xv = xb[c * 2400 + ((v < V2) ? v : 0)];  // clamped load
    Xs[c][v] = (__bf16)((v < V2) ? xv : 0.0f);
  }
  __syncthreads();

  const __bf16* abt = attw + (size_t)(n * NS) * (VP * KP);

  // A operands for GEMM1 (rows of X_t) — invariant over subset i, hoisted.
  v16bf Ax[3];
#pragma unroll
  for (int ch = 0; ch < 3; ++ch) {
    const int ab = 32 * ch + 16 * lhi;
    Ax[ch] = bfcat(ldv8(&Xs[16 * wave + l16][ab]),
                   ldv8(&Xs[16 * wave + l16][ab + 8]));
  }

  // GEMM1: V_i = X_t(64x96) @ att_i(96x80); att read transposed [u][k].
#pragma unroll
  for (int i = 0; i < NS; ++i) {
    v8f acc[5];
#pragma unroll
    for (int nt = 0; nt < 5; ++nt)
#pragma unroll
      for (int r = 0; r < 8; ++r) acc[nt][r] = 0.0f;
#pragma unroll
    for (int ch = 0; ch < 3; ++ch) {
      const int bb = 32 * ch + 16 * lhi;
#pragma unroll
      for (int nt = 0; nt < 5; ++nt) {
        const __bf16* bp = abt + (size_t)(i * VP + 16 * nt + l16) * KP + bb;
        v16bf Bt = bfcat(ldv8(bp), ldv8(bp + 8));  // 2x global_load_b128
        acc[nt] = wmma_bf16(Ax[ch], Bt, acc[nt]);
      }
    }
#pragma unroll
    for (int nt = 0; nt < 5; ++nt) {
      v8bf pv;
#pragma unroll
      for (int r = 0; r < 8; ++r) pv[r] = (__bf16)acc[nt][r];
      stv8(&Vst[16 * nt + l16][i * CCH + 16 * wave + 8 * lhi], pv);
    }
  }
  __syncthreads();

  // GEMM2: z = Wffn_big(64x192) @ V(192x80), V read transposed [u][k].
  v8f acc[5];
#pragma unroll
  for (int nt = 0; nt < 5; ++nt)
#pragma unroll
    for (int r = 0; r < 8; ++r) acc[nt][r] = 0.0f;
#pragma unroll
  for (int ch = 0; ch < 6; ++ch) {
    v16bf A;  // A[m=o][k=(i,c)] gathered from Wffn (L0/L2-resident)
#pragma unroll
    for (int e = 0; e < 16; ++e) {
      int k  = 32 * ch + 16 * lhi + e;  // 0..191
      int ii = k >> 6, c = k & 63;
      A[e] = (__bf16)Wffn[(ii * CCH + 16 * wave + l16) * CCH + c];
    }
    const int bb = 32 * ch + 16 * lhi;
#pragma unroll
    for (int nt = 0; nt < 5; ++nt) {
      v16bf Bt = bfcat(ldv8(&Vst[16 * nt + l16][bb]),
                       ldv8(&Vst[16 * nt + l16][bb + 8]));
      acc[nt] = wmma_bf16(A, Bt, acc[nt]);
    }
  }

  // Epilogue: bias + BN(eval) + residual + ReLU -> d_out.
  float* yb = y + (size_t)n * (CCH * 2400) + t * V2;
#pragma unroll
  for (int nt = 0; nt < 5; ++nt) {
    const int u = 16 * nt + l16;
    if (u < V2) {  // single exec toggle per N tile
#pragma unroll
      for (int r = 0; r < 8; ++r) {
        int o = 16 * wave + r + 8 * lhi;
        float z   = acc[nt][r] + bffn[o] + bffn[64 + o] + bffn[128 + o];
        float sc  = bn_g[o] * rsqrtf(bn_v[o] + 1e-5f);
        float val = (z - bn_m[o]) * sc + bn_b[o] + xb[o * 2400 + u];
        yb[o * 2400 + u] = fmaxf(val, 0.0f);
      }
    }
  }
}

// ---------------------------------------------------------------------------
// Kernel C: spatial (conv75) / temporal (conv9) / channel (SE MLP) attention,
// fused per batch sample; y updated in place (y *= sscale*tscale*cscale).
// ---------------------------------------------------------------------------
__global__ __launch_bounds__(256) void se_kernel(
    float* __restrict__ y,
    const float* __restrict__ w_sa, const float* __restrict__ b_sa,
    const float* __restrict__ w_ta, const float* __restrict__ b_ta,
    const float* __restrict__ w_fc1, const float* __restrict__ b_fc1,
    const float* __restrict__ w_fc2, const float* __restrict__ b_fc2) {
  __shared__ float sse[CCH][V2];
  __shared__ float sscale[V2];
  __shared__ float st[CCH][T2];
  __shared__ float tscale[T2];
  __shared__ float scpart[4][CCH];
  __shared__ float scmean[CCH];
  __shared__ float s1[32];
  __shared__ float cscale[CCH];
  const int n   = blockIdx.x;
  const int tid = threadIdx.x;
  float* yb = y + (size_t)n * (CCH * 2400);

  // spatial means over t
  for (int idx = tid; idx < CCH * V2; idx += 256) {
    int c = idx / V2, u = idx - c * V2;
    float s = 0.0f;
    for (int t = 0; t < T2; ++t) s += yb[c * 2400 + t * V2 + u];
    sse[c][u] = s * (1.0f / T2);
  }
  __syncthreads();
  // spatial conv (K=75, pad=37) + sigmoid -> 1+sig scale
  for (int u = tid; u < V2; u += 256) {
    float acc = b_sa[0];
    for (int c = 0; c < CCH; ++c) {
      const int j0 = (37 - u > 0) ? (37 - u) : 0;
      const int j1 = (V2 + 37 - u < V2) ? (V2 + 37 - u) : V2;
      for (int j = j0; j < j1; ++j)
        acc += w_sa[c * V2 + j] * sse[c][u + j - 37];
    }
    sscale[u] = 1.0f + sigm(acc);
  }
  __syncthreads();
  // temporal means over v (of spatially scaled y)
  for (int idx = tid; idx < CCH * T2; idx += 256) {
    int c = idx / T2, t = idx - c * T2;
    float s = 0.0f;
    for (int u = 0; u < V2; ++u) s += yb[c * 2400 + t * V2 + u] * sscale[u];
    st[c][t] = s * (1.0f / V2);
  }
  __syncthreads();
  // temporal conv (K=9, pad=4) + sigmoid
  for (int t = tid; t < T2; t += 256) {
    float acc = b_ta[0];
    for (int c = 0; c < CCH; ++c) {
      const int j0 = (4 - t > 0) ? (4 - t) : 0;
      const int j1 = (T2 + 4 - t < 9) ? (T2 + 4 - t) : 9;
      for (int j = j0; j < j1; ++j)
        acc += w_ta[c * 9 + j] * st[c][t + j - 4];
    }
    tscale[t] = 1.0f + sigm(acc);
  }
  __syncthreads();
  // channel means of doubly-scaled y (4-way partial over t)
  {
    int c = tid & 63, part = tid >> 6;
    float s = 0.0f;
    for (int t = part * 8; t < part * 8 + 8; ++t) {
      float ts = tscale[t];
      for (int u = 0; u < V2; ++u)
        s += yb[c * 2400 + t * V2 + u] * sscale[u] * ts;
    }
    scpart[part][c] = s;
  }
  __syncthreads();
  if (tid < CCH)
    scmean[tid] = (scpart[0][tid] + scpart[1][tid] + scpart[2][tid] +
                   scpart[3][tid]) * (1.0f / 2400.0f);
  __syncthreads();
  if (tid < 32) {
    float acc = b_fc1[tid];
    for (int c = 0; c < CCH; ++c) acc += w_fc1[tid * CCH + c] * scmean[c];
    s1[tid] = fmaxf(acc, 0.0f);
  }
  __syncthreads();
  if (tid < CCH) {
    float acc = b_fc2[tid];
    for (int h = 0; h < 32; ++h) acc += w_fc2[tid * 32 + h] * s1[h];
    cscale[tid] = 1.0f + sigm(acc);
  }
  __syncthreads();
  // final in-place scaling
  for (int idx = tid; idx < CCH * 2400; idx += 256) {
    int c = idx / 2400;
    int rem = idx - c * 2400;
    int t = rem / V2, u = rem - t * V2;
    yb[idx] = yb[idx] * (sscale[u] * tscale[t] * cscale[c]);
  }
}

// ---------------------------------------------------------------------------
extern "C" void kernel_launch(void* const* d_in, const int* in_sizes, int n_in,
                              void* d_out, int out_size, void* d_ws, size_t ws_size,
                              hipStream_t stream) {
  (void)n_in; (void)out_size; (void)ws_size;
  const float* x     = (const float*)d_in[0];
  const float* PA    = (const float*)d_in[1];
  const float* alpha = (const float*)d_in[2];
  const float* Wq    = (const float*)d_in[3];
  const float* bq    = (const float*)d_in[4];
  const float* Wk    = (const float*)d_in[5];
  const float* bk    = (const float*)d_in[6];
  const float* Wffn  = (const float*)d_in[7];
  const float* bffn  = (const float*)d_in[8];
  const float* bn_g  = (const float*)d_in[9];
  const float* bn_b  = (const float*)d_in[10];
  const float* bn_m  = (const float*)d_in[11];
  const float* bn_v  = (const float*)d_in[12];
  const float* w_sa  = (const float*)d_in[13];
  const float* b_sa  = (const float*)d_in[14];
  const float* w_ta  = (const float*)d_in[15];
  const float* b_ta  = (const float*)d_in[16];
  const float* w_fc1 = (const float*)d_in[17];
  const float* b_fc1 = (const float*)d_in[18];
  const float* w_fc2 = (const float*)d_in[19];
  const float* b_fc2 = (const float*)d_in[20];
  float* yout  = (float*)d_out;
  __bf16* attw = (__bf16*)d_ws;  // B*3*80*96 bf16 = ~5.9 MB

  const int B = in_sizes[0] / (CCH * 2400);  // 128

  attn_kernel<<<B * NS, 160, 0, stream>>>(x, PA, alpha, Wq, bq, Wk, bk, attw);
  gcn_kernel<<<B * T2, 128, 0, stream>>>(x, attw, Wffn, bffn, bn_g, bn_b, bn_m,
                                         bn_v, yout);
  se_kernel<<<B, 256, 0, stream>>>(yout, w_sa, b_sa, w_ta, b_ta, w_fc1, b_fc1,
                                   w_fc2, b_fc2);
}
